// LI_29609504539059
// MI455X (gfx1250) — matrix operand order repaired and checked by reading, not compiled, not run on hardware
//
#include <hip/hip_runtime.h>

typedef __attribute__((ext_vector_type(16))) __bf16 v16bf;
typedef __attribute__((ext_vector_type(8)))  float  v8f;

#define B_   32
#define T_   2048
#define IN_  1024
#define OUT_ 256

// ---------------------------------------------------------------------------
// Kernel 1: current = X @ W^T + bias, via bf16 WMMA with f32 accumulation.
//   X: [B*T, IN] f32 (row-major), W: [OUT, IN] f32 (row-major)
//   Y: [B*T, OUT] f32
// One wave (32 lanes) computes one 16x16 tile of Y. 8 waves per block share
// the same ntile (W rows reused); grid.x = ntile so blocks with the same
// mtile-group reuse X rows via L2 (192 MB).
// ---------------------------------------------------------------------------
__global__ __launch_bounds__(256) void li_gemm_wmma_bf16(
    const float* __restrict__ X,
    const float* __restrict__ W,
    const float* __restrict__ bias,
    float* __restrict__ Y)
{
    const int lane  = threadIdx.x & 31;
    const int wave  = threadIdx.x >> 5;
    const int ntile = blockIdx.x;                 // 0..15   (OUT/16)
    const int mtile = blockIdx.y * 8 + wave;      // 0..4095 (B*T/16)

    const int half = lane >> 4;                   // 0 or 1
    const int mn   = lane & 15;                   // M for A, N for B/C/D

    const float* __restrict__ xrow = X + (size_t)(mtile * 16 + mn) * IN_;
    const float* __restrict__ wrow = W + (size_t)(ntile * 16 + mn) * IN_;

    v8f c = {};

    #pragma unroll 4
    for (int kc = 0; kc < IN_ / 32; ++kc) {
        const int kb = kc * 32;

        // Prefetch next X chunk (gfx1250 global_prefetch path).
        if (kc + 1 < IN_ / 32) {
            __builtin_prefetch(xrow + kb + 32, 0, 3);
        }

        // ---- A fragment: 16x32 bf16 (M x K), contiguous K (matches B) ----
        //  lanes 0-15 : K = kb+0..15 ; lanes 16-31: K = kb+16..31
        const float* a0 = xrow + kb + 16 * half;       // group K = 16*half .. +7
        const float* a1 = xrow + kb + 16 * half + 8;   // group K = 16*half+8 .. +7
        float af[16];
        *(float4*)(af + 0)  = *(const float4*)(a0);
        *(float4*)(af + 4)  = *(const float4*)(a0 + 4);
        *(float4*)(af + 8)  = *(const float4*)(a1);
        *(float4*)(af + 12) = *(const float4*)(a1 + 4);

        // ---- B fragment: 32x16 bf16 (K x N), B[k][n] = W[n][k] ----
        //  lanes 0-15 : K = kb+0..15 ; lanes 16-31: K = kb+16..31
        const float* bp = wrow + kb + 16 * half;
        float bf[16];
        *(float4*)(bf + 0)  = *(const float4*)(bp);
        *(float4*)(bf + 4)  = *(const float4*)(bp + 4);
        *(float4*)(bf + 8)  = *(const float4*)(bp + 8);
        *(float4*)(bf + 12) = *(const float4*)(bp + 12);

        v16bf av, bv;
        #pragma unroll
        for (int j = 0; j < 16; ++j) {
            av[j] = (__bf16)af[j];
            bv[j] = (__bf16)bf[j];
        }

        // D = A x B + C   (f32 accumulate)
        c = __builtin_amdgcn_wmma_f32_16x16x32_bf16(
                /*neg_a=*/false, av, /*neg_b=*/false, bv,
                /*c_mod=*/(short)0, c, /*reuse_a=*/false, /*reuse_b=*/false);
    }

    // ---- C/D layout: VGPR r, lanes 0-15 -> M=r, lanes 16-31 -> M=r+8 ----
    const float badd = bias[ntile * 16 + mn];
    float* __restrict__ yb =
        Y + (size_t)(mtile * 16 + 8 * half) * OUT_ + ntile * 16 + mn;
    #pragma unroll
    for (int r = 0; r < 8; ++r) {
        yb[(size_t)r * OUT_] = c[r] + badd;
    }
}

// ---------------------------------------------------------------------------
// Kernel 2: in-place leaky-integrator scan over t for each (b, o) channel.
//   u_t = alpha*u_{t-1} + (1-alpha)*x_t, u_0 = 0
// Consecutive o across lanes -> fully coalesced 1 KiB rows per step.
// ---------------------------------------------------------------------------
__global__ __launch_bounds__(256) void li_scan(
    float* __restrict__ Y, const float* __restrict__ decay)
{
    const int idx = blockIdx.x * blockDim.x + threadIdx.x; // 0..B*OUT-1
    const int b = idx / OUT_;
    const int o = idx % OUT_;

    const float alpha = decay[o];
    const float omin  = 1.0f - alpha;

    float* __restrict__ p = Y + (size_t)b * T_ * OUT_ + o;
    float u = 0.0f;
    for (int t = 0; t < T_; ++t) {
        const float x = p[(size_t)t * OUT_];
        u = alpha * u + omin * x;
        p[(size_t)t * OUT_] = u;
    }
}

extern "C" void kernel_launch(void* const* d_in, const int* in_sizes, int n_in,
                              void* d_out, int out_size, void* d_ws, size_t ws_size,
                              hipStream_t stream) {
    (void)in_sizes; (void)n_in; (void)out_size; (void)d_ws; (void)ws_size;

    const float* X     = (const float*)d_in[0];   // [B,T,IN]
    const float* W     = (const float*)d_in[1];   // [OUT,IN]
    const float* bias  = (const float*)d_in[2];   // [OUT]
    const float* decay = (const float*)d_in[3];   // [OUT]
    float*       Y     = (float*)d_out;           // [B,T,OUT]

    // GEMM: 16 ntiles x (4096 mtiles / 8 waves per block)
    dim3 g1(OUT_ / 16, (B_ * T_ / 16) / 8);
    li_gemm_wmma_bf16<<<g1, 256, 0, stream>>>(X, W, bias, Y);

    // Scan: B*OUT = 8192 channels
    dim3 g2((B_ * OUT_) / 256);
    li_scan<<<g2, 256, 0, stream>>>(Y, decay);
}
